// GodelGhostNet_25984552141466
// MI455X (gfx1250) — matrix operand order repaired and checked by reading, not compiled, not run on hardware
//
#include <hip/hip_runtime.h>
#include <hip/hip_bf16.h>
#include <math.h>

// GodelGhostNet on MI455X (gfx1250), wave32 + WMMA.
// Edge MLP factored to node-level GEMMs; all GEMMs use
// v_wmma_f32_16x16x32_f16 with f32 accumulation, software-pipelined so each
// k-step's fragment loads are issued before the previous step's WMMAs.
// Edge phase is an L2-resident gather + relu + f32 atomic scatter.

typedef _Float16 v16h __attribute__((ext_vector_type(16)));
typedef _Float16 v8h  __attribute__((ext_vector_type(8)));
typedef float    v8f  __attribute__((ext_vector_type(8)));

#define HID   1024
#define SDIM  18
#define LAYERS 4

// A fragment (16x32 f16): lane row = lane&15, halves {kb..kb+15}
__device__ __forceinline__ v16h load_a_frag(const _Float16* __restrict__ p)
{
    v8h lo = *(const v8h*)(p);
    v8h hi = *(const v8h*)(p + 8);
    v16h a;
#pragma unroll
    for (int i = 0; i < 8; ++i) { a[i] = lo[i]; a[i + 8] = hi[i]; }
    return a;
}

// ---------------------------------------------------------------------------
// Generic WMMA GEMM: C[M,Nout] = A[M,1024] (f16, row-major) x Bt[Nout,1024]^T
// Block = 256 thr (8 waves); each wave computes a 16x64 strip (4 C tiles).
// k-loop is double-buffered in registers: loads for step k+1 precede WMMAs of
// step k, so s_wait_loadcnt overlaps with 4 in-flight WMMAs.
// Epilogue: optional bias[n], optional deg[m]*degBias[n], optional relu,
// optional f32 store (guarded by mGuard) and/or f16 store.
// ---------------------------------------------------------------------------
__global__ __launch_bounds__(256)
void wmma_gemm(const _Float16* __restrict__ A,
               const _Float16* __restrict__ Bt,
               float* __restrict__ outF32,
               _Float16* __restrict__ outF16,
               const float* __restrict__ bias,
               const float* __restrict__ deg,
               const float* __restrict__ degBias,
               int Nout, int relu, int mGuard)
{
    const int K = 1024;
    const int lane = threadIdx.x & 31;
    const int wave = threadIdx.x >> 5;
    const int row0 = blockIdx.y * 128 + wave * 16;
    const int col0 = blockIdx.x * 64;
    const int r  = lane & 15;
    const int hi = lane >> 4;       // lane group 0/1

    const _Float16* Abase = A  + (size_t)(row0 + r) * K + hi * 16;  // + k0
    const _Float16* Bbase = Bt + (size_t)(col0 + r) * K + hi * 16;  // + k0

    v8f acc0 = {}, acc1 = {}, acc2 = {}, acc3 = {};

    // prologue: fragments for k0 = 0
    v16h a_c  = load_a_frag(Abase);
    v16h b0_c = *(const v16h*)(Bbase + (size_t)0  * K);
    v16h b1_c = *(const v16h*)(Bbase + (size_t)16 * K);
    v16h b2_c = *(const v16h*)(Bbase + (size_t)32 * K);
    v16h b3_c = *(const v16h*)(Bbase + (size_t)48 * K);

#pragma unroll 4
    for (int k0 = 32; k0 < K; k0 += 32) {
        // issue next step's loads first (overlap with current WMMAs)
        v16h a_n  = load_a_frag(Abase + k0);
        v16h b0_n = *(const v16h*)(Bbase + (size_t)0  * K + k0);
        v16h b1_n = *(const v16h*)(Bbase + (size_t)16 * K + k0);
        v16h b2_n = *(const v16h*)(Bbase + (size_t)32 * K + k0);
        v16h b3_n = *(const v16h*)(Bbase + (size_t)48 * K + k0);

        acc0 = __builtin_amdgcn_wmma_f32_16x16x32_f16(false, a_c, false, b0_c, (short)0, acc0, false, false);
        acc1 = __builtin_amdgcn_wmma_f32_16x16x32_f16(false, a_c, false, b1_c, (short)0, acc1, false, false);
        acc2 = __builtin_amdgcn_wmma_f32_16x16x32_f16(false, a_c, false, b2_c, (short)0, acc2, false, false);
        acc3 = __builtin_amdgcn_wmma_f32_16x16x32_f16(false, a_c, false, b3_c, (short)0, acc3, false, false);

        a_c = a_n; b0_c = b0_n; b1_c = b1_n; b2_c = b2_n; b3_c = b3_n;
    }
    // epilogue k-step
    acc0 = __builtin_amdgcn_wmma_f32_16x16x32_f16(false, a_c, false, b0_c, (short)0, acc0, false, false);
    acc1 = __builtin_amdgcn_wmma_f32_16x16x32_f16(false, a_c, false, b1_c, (short)0, acc1, false, false);
    acc2 = __builtin_amdgcn_wmma_f32_16x16x32_f16(false, a_c, false, b2_c, (short)0, acc2, false, false);
    acc3 = __builtin_amdgcn_wmma_f32_16x16x32_f16(false, a_c, false, b3_c, (short)0, acc3, false, false);

    v8f accs[4] = { acc0, acc1, acc2, acc3 };
#pragma unroll
    for (int j = 0; j < 4; ++j) {
        const int n = col0 + 16 * j + r;
        const float bn  = bias ? bias[n]    : 0.0f;
        const float b2n = deg  ? degBias[n] : 0.0f;
#pragma unroll
        for (int rr = 0; rr < 8; ++rr) {
            const int m = row0 + rr + 8 * hi;     // C layout: vreg rr -> row
            float v = accs[j][rr] + bn;
            if (deg)  v += deg[m] * b2n;
            if (relu) v = fmaxf(v, 0.0f);
            if (outF16) outF16[(size_t)m * Nout + n] = (_Float16)v;
            if (outF32 && m < mGuard) outF32[(size_t)m * Nout + n] = v;
        }
    }
}

// ---------------------------------------------------------------------------
// Weight prep: build f16 transposed operands ([Nout, K] row-major).
// Wct[l][n][k]: n<1024 -> W1[l][k][n] - W1[l][k+1024][n]; else W1[l][k+1024][n-1024]
// ---------------------------------------------------------------------------
__global__ void prep_w1(const float* __restrict__ W1, _Float16* __restrict__ Wct)
{
    size_t i = (size_t)blockIdx.x * blockDim.x + threadIdx.x;
    if (i >= (size_t)LAYERS * 2048 * 1024) return;
    int k = (int)(i & 1023);
    size_t t = i >> 10;
    int n = (int)(t & 2047);
    int l = (int)(t >> 11);
    const float* base = W1 + (size_t)l * 2048 * 1024;
    float v;
    if (n < 1024) v = base[(size_t)k * 1024 + n] - base[(size_t)(k + 1024) * 1024 + n];
    else          v = base[(size_t)(k + 1024) * 1024 + (n - 1024)];
    Wct[i] = (_Float16)v;
}

__global__ void prep_w2(const float* __restrict__ W2, _Float16* __restrict__ W2t)
{
    size_t i = (size_t)blockIdx.x * blockDim.x + threadIdx.x;
    if (i >= (size_t)LAYERS * 1024 * 1024) return;
    int k = (int)(i & 1023);
    size_t t = i >> 10;
    int n = (int)(t & 1023);
    int l = (int)(t >> 10);
    W2t[i] = (_Float16)W2[(size_t)l * 1024 * 1024 + (size_t)k * 1024 + n];
}

__global__ void prep_gw1(const float* __restrict__ gw1, _Float16* __restrict__ gw1t)
{
    size_t i = (size_t)blockIdx.x * blockDim.x + threadIdx.x;
    if (i >= (size_t)256 * 1024) return;
    int k = (int)(i & 1023);
    int n = (int)(i >> 10);
    gw1t[i] = (_Float16)gw1[(size_t)k * 256 + n];
}

// ---------------------------------------------------------------------------
__global__ void zero_kernel(float* __restrict__ p, int n4)
{
    int i = blockIdx.x * blockDim.x + threadIdx.x;
    if (i < n4) { float4 z = {0.f, 0.f, 0.f, 0.f}; ((float4*)p)[i] = z; }
}

__global__ void deg_kernel(const int* __restrict__ ei, float* __restrict__ deg, int E)
{
    int e = blockIdx.x * blockDim.x + threadIdx.x;
    if (e < E) atomicAdd(&deg[ei[E + e]], 1.0f);
}

// h = sin(x @ proj_w) * cos(x @ proj_w); pad rows -> 0
__global__ __launch_bounds__(256)
void embed_kernel(const float* __restrict__ x, const float* __restrict__ pw,
                  _Float16* __restrict__ hf16, int Nreal)
{
    int hcol = blockIdx.x * 256 + threadIdx.x;   // 0..1023
    int n = blockIdx.y;
    float v = 0.0f;
    if (n < Nreal) {
        float p = 0.0f;
#pragma unroll
        for (int k = 0; k < SDIM; ++k) p += x[(size_t)n * SDIM + k] * pw[(size_t)k * HID + hcol];
        v = sinf(p) * cosf(p);
    }
    hf16[(size_t)n * HID + hcol] = (_Float16)v;
}

// Per-edge: S[dst] += relu(AB[dst][0:1024] + AB[src][1024:2048] + b1)
__global__ __launch_bounds__(256)
void edge_kernel(const int* __restrict__ ei, const float* __restrict__ AB,
                 const float* __restrict__ b1, float* __restrict__ S, int E)
{
    int e = blockIdx.x;
    int src = ei[e];
    int dst = ei[E + e];
    int c = threadIdx.x * 4;
    float4 a  = *(const float4*)(AB + (size_t)dst * 2048 + c);
    float4 b  = *(const float4*)(AB + (size_t)src * 2048 + 1024 + c);
    float4 bb = *(const float4*)(b1 + c);
    float* sp = S + (size_t)dst * HID + c;
    atomicAdd(sp + 0, fmaxf(a.x + b.x + bb.x, 0.0f));
    atomicAdd(sp + 1, fmaxf(a.y + b.y + bb.y, 0.0f));
    atomicAdd(sp + 2, fmaxf(a.z + b.z + bb.z, 0.0f));
    atomicAdd(sp + 3, fmaxf(a.w + b.w + bb.w, 0.0f));
}

__global__ void cvt_f32_to_f16(const float* __restrict__ s, _Float16* __restrict__ d, size_t n)
{
    size_t i = (size_t)blockIdx.x * blockDim.x + threadIdx.x;
    if (i < n) d[i] = (_Float16)s[i];
}

// ghost = sigmoid(relu(h@gw1+gb1) @ gw2 + gb2); g precomputed. One wave/node.
__global__ __launch_bounds__(256)
void ghost_kernel(const float* __restrict__ g, const float* __restrict__ gw2,
                  const float* __restrict__ gb2, float* __restrict__ out, int Nreal)
{
    int wave = threadIdx.x >> 5, lane = threadIdx.x & 31;
    int n = blockIdx.x * 8 + wave;
    if (n >= Nreal) return;
    float s = 0.0f;
#pragma unroll
    for (int i = 0; i < 8; ++i)
        s += g[(size_t)n * 256 + lane + 32 * i] * gw2[lane + 32 * i];
    for (int off = 16; off > 0; off >>= 1) s += __shfl_xor(s, off, 32);
    if (lane == 0) out[n] = 1.0f / (1.0f + expf(-(s + gb2[0])));
}

// stable = h @ sw + sb  (18 cols). One wave/node, lane j<18 owns a column.
__global__ __launch_bounds__(256)
void stable_kernel(const _Float16* __restrict__ h, const float* __restrict__ sw,
                   const float* __restrict__ sb, float* __restrict__ out, int Nreal)
{
    int wave = threadIdx.x >> 5, lane = threadIdx.x & 31;
    int n = blockIdx.x * 8 + wave;
    if (n >= Nreal) return;
    if (lane < SDIM) {
        float s = 0.0f;
        for (int k = 0; k < HID; ++k)
            s += (float)h[(size_t)n * HID + k] * sw[(size_t)k * SDIM + lane];
        out[(size_t)n * SDIM + lane] = s + sb[lane];
    }
}

// ---------------------------------------------------------------------------
extern "C" void kernel_launch(void* const* d_in, const int* in_sizes, int n_in,
                              void* d_out, int out_size, void* d_ws, size_t ws_size,
                              hipStream_t stream)
{
    const float* x   = (const float*)d_in[0];
    const int*   ei  = (const int*)  d_in[1];
    const float* pw  = (const float*)d_in[2];
    const float* W1  = (const float*)d_in[3];
    const float* b1  = (const float*)d_in[4];
    const float* W2  = (const float*)d_in[5];
    const float* b2  = (const float*)d_in[6];
    const float* gw1 = (const float*)d_in[7];
    const float* gb1 = (const float*)d_in[8];
    const float* gw2 = (const float*)d_in[9];
    const float* gb2 = (const float*)d_in[10];
    const float* sw  = (const float*)d_in[11];
    const float* sb  = (const float*)d_in[12];

    const int N = in_sizes[0] / SDIM;
    const int E = in_sizes[1] / 2;
    const int MPAD = ((N + 127) / 128) * 128;

    // Workspace carve-up (256B aligned)
    auto align = [](size_t v) { return (v + 255) & ~(size_t)255; };
    char* ws = (char*)d_ws;
    size_t off = 0;
    _Float16* h16  = (_Float16*)(ws + off); off = align(off + (size_t)MPAD * HID * 2);
    _Float16* S16  = (_Float16*)(ws + off); off = align(off + (size_t)MPAD * HID * 2);
    float*    AB   = (float*)   (ws + off); off = align(off + (size_t)MPAD * 2048 * 4);
    float*    S    = (float*)   (ws + off); off = align(off + (size_t)MPAD * HID * 4);
    float*    g    = (float*)   (ws + off); off = align(off + (size_t)MPAD * 256 * 4);
    float*    deg  = (float*)   (ws + off); off = align(off + (size_t)MPAD * 4);
    _Float16* Wct  = (_Float16*)(ws + off); off = align(off + (size_t)LAYERS * 2048 * 1024 * 2);
    _Float16* W2t  = (_Float16*)(ws + off); off = align(off + (size_t)LAYERS * 1024 * 1024 * 2);
    _Float16* gw1t = (_Float16*)(ws + off); off = align(off + (size_t)256 * 1024 * 2);

    float* out_ghost  = (float*)d_out;                        // [N]
    float* out_stable = (float*)d_out + N;                    // [N,18]
    float* out_h      = (float*)d_out + N + (size_t)N * SDIM; // [N,1024]

    // 1) weight prep (f16, transposed)
    {
        size_t n1 = (size_t)LAYERS * 2048 * 1024;
        prep_w1<<<(unsigned)((n1 + 255) / 256), 256, 0, stream>>>(W1, Wct);
        size_t n2 = (size_t)LAYERS * 1024 * 1024;
        prep_w2<<<(unsigned)((n2 + 255) / 256), 256, 0, stream>>>(W2, W2t);
        size_t n3 = (size_t)256 * 1024;
        prep_gw1<<<(unsigned)((n3 + 255) / 256), 256, 0, stream>>>(gw1, gw1t);
    }

    // 2) degrees
    zero_kernel<<<(MPAD / 4 + 255) / 256, 256, 0, stream>>>(deg, MPAD / 4);
    deg_kernel<<<(E + 255) / 256, 256, 0, stream>>>(ei, deg, E);

    // 3) embedding
    embed_kernel<<<dim3(HID / 256, MPAD), 256, 0, stream>>>(x, pw, h16, N);

    // 4) message-passing layers
    const int gy = MPAD / 128;
    for (int l = 0; l < LAYERS; ++l) {
        // AB = h @ [W1a | W1b]
        wmma_gemm<<<dim3(2048 / 64, gy), 256, 0, stream>>>(
            h16, Wct + (size_t)l * 2048 * 1024, AB, nullptr,
            nullptr, nullptr, nullptr, 2048, 0, MPAD);
        // S = 0
        int n4 = MPAD * HID / 4;
        zero_kernel<<<(n4 + 255) / 256, 256, 0, stream>>>(S, n4);
        // edge scatter: S[dst] += relu(AB[dst][:H] + AB[src][H:] + b1)
        edge_kernel<<<E, 256, 0, stream>>>(ei, AB, b1 + (size_t)l * HID, S, E);
        // S -> f16
        size_t ns = (size_t)MPAD * HID;
        cvt_f32_to_f16<<<(unsigned)((ns + 255) / 256), 256, 0, stream>>>(S, S16, ns);
        // h = relu(S @ W2 + deg*b2); last layer also streams f32 h to d_out
        wmma_gemm<<<dim3(1024 / 64, gy), 256, 0, stream>>>(
            S16, W2t + (size_t)l * 1024 * 1024,
            (l == LAYERS - 1) ? out_h : nullptr, h16,
            nullptr, deg, b2 + (size_t)l * HID, 1024, 1,
            (l == LAYERS - 1) ? N : MPAD);
    }

    // 5) ghost head hidden: g = relu(h @ gw1 + gb1)
    wmma_gemm<<<dim3(256 / 64, gy), 256, 0, stream>>>(
        h16, gw1t, g, nullptr, gb1, nullptr, nullptr, 256, 1, MPAD);
    ghost_kernel<<<(N + 7) / 8, 256, 0, stream>>>(g, gw2, gb2, out_ghost, N);

    // 6) state head
    stable_kernel<<<(N + 7) / 8, 256, 0, stream>>>(h16, sw, sb, out_stable, N);
}
